// MultiScaleAttention_81063212744881
// MI455X (gfx1250) — matrix operand order repaired and checked by reading, not compiled, not run on hardware
//
#include <hip/hip_runtime.h>
#include <hip/hip_bf16.h>

// ---------------------------------------------------------------------------
// MultiScaleAttention forward for MI455X (gfx1250), wave32 + WMMA bf16.
//   q = (x@wq^T + bq)/8   k = x@wk^T+bk   v = x@wv^T+bv   (bf16 WMMA GEMMs)
//   attn = softmax(q k^T) v                              (flash-style, WMMA)
//   gattn = ctx@wo^T + bo                                (bf16 WMMA GEMM)
//   out = LN(x + gattn + 0.3*depthwise_conv3(x))         (fp32 VALU)
// Staging: double-buffered gfx1250 async loads (ASYNCcnt, in-order retire).
// ---------------------------------------------------------------------------

typedef __bf16 bf16_t;
typedef __attribute__((ext_vector_type(8)))  __bf16 v8bf;
typedef __attribute__((ext_vector_type(16))) __bf16 v16bf;
typedef __attribute__((ext_vector_type(8)))  float  v8f;

#define D_MODEL 1024
#define NHEAD   16
#define D_K_    64
#define BATCH   4
#define SEQ     2048
#define LN_EPS  1e-5f

#define OUT_BHSD   0   // bf16 [B,H,S,Dk]
#define OUT_BHDS   1   // bf16 [B,H,Dk,S]  (pre-transposed V)
#define OUT_F32_RM 2   // fp32 row-major [M,N]

union BF16x16 { v16bf v; v8bf h[2]; };

static __device__ __forceinline__ v16bf join8(v8bf lo, v8bf hi) {
  BF16x16 u; u.h[0] = lo; u.h[1] = hi; return u.v;
}

// gfx1250 async copy: 16B global -> LDS, tracked by ASYNCcnt.
// Low 32 bits of a generic shared pointer are the LDS byte offset
static __device__ __forceinline__ void async_copy_b128(bf16_t* lds_dst,
                                                       const bf16_t* gsrc) {
  unsigned ldsa = (unsigned)(uintptr_t)lds_dst;
  asm volatile("global_load_async_to_lds_b128 %0, %1, off"
               :: "v"(ldsa), "v"(gsrc) : "memory");
}
static __device__ __forceinline__ void async_wait0() {
  asm volatile("s_wait_asynccnt 0x0" ::: "memory");
}
// Async loads retire in order: waiting <=4 with 8 outstanding retires the
// older tile's 4 copies while the newer tile's 4 stay in flight.
static __device__ __forceinline__ void async_wait4() {
  asm volatile("s_wait_asynccnt 0x4" ::: "memory");
}

// ---------------------------------------------------------------------------
__global__ __launch_bounds__(256) void f32_to_bf16_kernel(
    const float* __restrict__ in, bf16_t* __restrict__ out, int n) {
  int i = blockIdx.x * blockDim.x + threadIdx.x;
  int stride = gridDim.x * blockDim.x;
  for (; i < n; i += stride) out[i] = (bf16_t)in[i];
}

// ---------------------------------------------------------------------------
// C[M,N] = (A[M,K] @ W[N,K]^T + bias) * out_scale, A/W bf16, acc fp32.
// Block: 256 thr (8 waves) -> 128x128 C tile; wave grid 4(M)x2(N), each wave
// 32x64 = 8 accumulators -> 8 independent WMMAs per 32-deep K step.
// Double-buffered async staging overlaps next tile load with WMMA burst.
// ---------------------------------------------------------------------------
template <int MODE>
__global__ __launch_bounds__(256) void gemm_xwT_kernel(
    const bf16_t* __restrict__ A, const bf16_t* __restrict__ W,
    const float* __restrict__ bias, void* __restrict__ outp,
    int M, int N, int Kdim, float out_scale)
{
  __shared__ __align__(16) bf16_t As[2][128 * 40];   // [row][k], pad 40 (80B)
  __shared__ __align__(16) bf16_t Bs[2][128 * 40];
  const int tid  = threadIdx.x;
  const int wave = tid >> 5, lane = tid & 31;
  const int lm   = lane & 15, half = lane >> 4;
  const int mw   = wave & 3;        // 4 M subtiles of 32 rows
  const int nw   = wave >> 2;       // 2 N subtiles of 64 cols
  const int mbase = blockIdx.y * 128;
  const int nbase = blockIdx.x * 128;

  auto stage = [&](int kb, bf16_t* Ab, bf16_t* Bb) {
#pragma unroll
    for (int i = 0; i < 2; ++i) {
      int idx = tid + i * 256;
      int row = idx >> 2, chunk = idx & 3;
      async_copy_b128(Ab + row * 40 + chunk * 8,
                      A + (size_t)(mbase + row) * Kdim + kb * 32 + chunk * 8);
      async_copy_b128(Bb + row * 40 + chunk * 8,
                      W + (size_t)(nbase + row) * Kdim + kb * 32 + chunk * 8);
    }
  };

  v8f zero = {};
  v8f acc[2][4];
#pragma unroll
  for (int i = 0; i < 2; ++i)
#pragma unroll
    for (int j = 0; j < 4; ++j) acc[i][j] = zero;

  const int ksteps = Kdim >> 5;
  stage(0, As[0], Bs[0]);                     // prologue: tile 0 in flight
  for (int kb = 0; kb < ksteps; ++kb) {
    const int cur = kb & 1;
    __syncthreads();                          // buf[cur^1] readers done
    if (kb + 1 < ksteps) {
      stage(kb + 1, As[cur ^ 1], Bs[cur ^ 1]);
      async_wait4();                          // tile kb retired, kb+1 in flight
    } else {
      async_wait0();
    }
    __syncthreads();                          // tile kb visible to all waves

    // A fragments: rows mw*32 + {0,16} + lm; K runs [8h..] and [16+8h..]
    v16bf af[2];
#pragma unroll
    for (int i = 0; i < 2; ++i) {
      const bf16_t* ap = As[cur] + (mw * 32 + i * 16 + lm) * 40;
      af[i] = join8(*(const v8bf*)(ap + 8 * half),
                    *(const v8bf*)(ap + 16 + 8 * half));
    }
    // B fragments: cols nw*64 + {0,16,32,48} + lm; K runs [8h..] and [16+8h..]
    v16bf bfm[4];
#pragma unroll
    for (int j = 0; j < 4; ++j) {
      const bf16_t* bp = Bs[cur] + (nw * 64 + j * 16 + lm) * 40 + 8 * half;
      bfm[j] = join8(*(const v8bf*)bp, *(const v8bf*)(bp + 16));
    }
#pragma unroll
    for (int i = 0; i < 2; ++i)
#pragma unroll
      for (int j = 0; j < 4; ++j)
        acc[i][j] = __builtin_amdgcn_wmma_f32_16x16x32_bf16(
            false, af[i], false, bfm[j], (short)0, acc[i][j], false, false);
  }

  // Epilogue: C element r <-> row (r + 8*half), col lm. All index math is
  // hoisted; the whole 128-row block lives in one batch (2048 % 128 == 0).
  const int bi = mbase / SEQ;
#pragma unroll
  for (int i = 0; i < 2; ++i) {
#pragma unroll
    for (int j = 0; j < 4; ++j) {
      const int gn  = nbase + nw * 64 + j * 16 + lm;
      const int gm0 = mbase + mw * 32 + i * 16 + 8 * half;  // row of r=0
      const float bv = bias[gn];
      if (MODE == OUT_F32_RM) {
        float* p = (float*)outp + (size_t)gm0 * N + gn;
#pragma unroll
        for (int r = 0; r < 8; ++r)
          p[(size_t)r * N] = (acc[i][j][r] + bv) * out_scale;
      } else {
        const int h = gn >> 6, dk = gn & 63;
        const int s0 = gm0 - bi * SEQ;
        if (MODE == OUT_BHSD) {
          bf16_t* p = (bf16_t*)outp +
                      (((size_t)(bi * NHEAD + h) * SEQ) + s0) * D_K_ + dk;
#pragma unroll
          for (int r = 0; r < 8; ++r)
            p[r * D_K_] = (bf16_t)((acc[i][j][r] + bv) * out_scale);
        } else {  // OUT_BHDS: unit stride along s
          bf16_t* p = (bf16_t*)outp +
                      (((size_t)(bi * NHEAD + h) * D_K_) + dk) * SEQ + s0;
#pragma unroll
          for (int r = 0; r < 8; ++r)
            p[r] = (bf16_t)((acc[i][j][r] + bv) * out_scale);
        }
      }
    }
  }
}

// ---------------------------------------------------------------------------
// Flash attention: block = (b,h,128 queries), 8 waves x 16-row q tiles.
// 64-key steps: 8 WMMA (scores 16x64) + online softmax + 8 WMMA (P@V).
// K/V^T staging double-buffered through the async engine.
// ---------------------------------------------------------------------------
__global__ __launch_bounds__(256) void attn_kernel(
    const bf16_t* __restrict__ Q, const bf16_t* __restrict__ Km,
    const bf16_t* __restrict__ Vt, bf16_t* __restrict__ ctx)
{
  __shared__ __align__(16) bf16_t Ks[2][64 * 72];    // [key][dk],  pad 72
  __shared__ __align__(16) bf16_t VTs[2][64 * 72];   // [dk][key],  pad 72
  __shared__ __align__(16) bf16_t Ps[8 * 16 * 72];   // per-wave P relayout
  const int tid  = threadIdx.x;
  const int wave = tid >> 5, lane = tid & 31;
  const int lm   = lane & 15, half = lane >> 4;
  const int qblk = blockIdx.x, h = blockIdx.y, b = blockIdx.z;
  const int bh   = b * NHEAD + h;
  const int qbase = qblk * 128 + wave * 16;

  auto stage = [&](int it, bf16_t* Kb, bf16_t* Vb) {
#pragma unroll
    for (int i = 0; i < 2; ++i) {
      int idx = tid + i * 256;
      int row = idx >> 3, chunk = idx & 7;
      async_copy_b128(Kb + row * 72 + chunk * 8,
                      Km + (((size_t)bh * SEQ) + it * 64 + row) * D_K_ + chunk * 8);
      async_copy_b128(Vb + row * 72 + chunk * 8,
                      Vt + (((size_t)bh * D_K_) + row) * SEQ + it * 64 + chunk * 8);
    }
  };

  // Q fragments (1/8 scale pre-folded): dk 0..31 and 32..63
  const bf16_t* qrow = Q + ((size_t)bh * SEQ + qbase + lm) * D_K_;
  v16bf qa[2];
  qa[0] = join8(*(const v8bf*)(qrow + 8 * half), *(const v8bf*)(qrow + 16 + 8 * half));
  qa[1] = join8(*(const v8bf*)(qrow + 32 + 8 * half), *(const v8bf*)(qrow + 48 + 8 * half));

  float m_run[8], l_run[8];
  v8f zero = {};
  v8f o[4];
#pragma unroll
  for (int r = 0; r < 8; ++r) { m_run[r] = -1e30f; l_run[r] = 0.f; }
#pragma unroll
  for (int d = 0; d < 4; ++d) o[d] = zero;

  bf16_t* Pw = Ps + wave * (16 * 72);

  const int iters = SEQ / 64;
  stage(0, Ks[0], VTs[0]);                    // prologue: tile 0 in flight
  for (int it = 0; it < iters; ++it) {
    const int cur = it & 1;
    __syncthreads();                          // buf[cur^1] readers done
    if (it + 1 < iters) {
      stage(it + 1, Ks[cur ^ 1], VTs[cur ^ 1]);
      async_wait4();
    } else {
      async_wait0();
    }
    __syncthreads();

    // scores: 16 queries x 64 keys = four 16x16 C tiles, 2 WMMAs each (dk)
    v8f sc[4];
#pragma unroll
    for (int ct = 0; ct < 4; ++ct) {
      v8f s = zero;
#pragma unroll
      for (int d = 0; d < 2; ++d) {
        const bf16_t* bp = Ks[cur] + (ct * 16 + lm) * 72 + d * 32 + 8 * half;
        v16bf bfrag = join8(*(const v8bf*)bp, *(const v8bf*)(bp + 16));
        s = __builtin_amdgcn_wmma_f32_16x16x32_bf16(
            false, qa[d], false, bfrag, (short)0, s, false, false);
      }
      sc[ct] = s;
    }

    // online softmax; C row = r + 8*half, col = lm (reduce in 16-lane group)
#pragma unroll
    for (int r = 0; r < 8; ++r) {
      float rm = fmaxf(fmaxf(sc[0][r], sc[1][r]), fmaxf(sc[2][r], sc[3][r]));
#pragma unroll
      for (int off = 1; off < 16; off <<= 1) rm = fmaxf(rm, __shfl_xor(rm, off, 32));
      float mnew  = fmaxf(m_run[r], rm);
      float scale = __expf(m_run[r] - mnew);
      float p[4];
      float rs = 0.f;
#pragma unroll
      for (int ct = 0; ct < 4; ++ct) { p[ct] = __expf(sc[ct][r] - mnew); rs += p[ct]; }
#pragma unroll
      for (int off = 1; off < 16; off <<= 1) rs += __shfl_xor(rs, off, 32);
      l_run[r] = l_run[r] * scale + rs;
      m_run[r] = mnew;
#pragma unroll
      for (int d = 0; d < 4; ++d) o[d][r] = o[d][r] * scale;
      int prow = r + 8 * half;
#pragma unroll
      for (int ct = 0; ct < 4; ++ct)
        Pw[prow * 72 + ct * 16 + lm] = (bf16_t)p[ct];   // C -> A relayout
    }

    // P fragments (16x64 -> two 16x32 A frags); same-wave LDS is in-order
    v16bf pa[2];
#pragma unroll
    for (int c = 0; c < 2; ++c)
      pa[c] = join8(*(const v8bf*)(Pw + lm * 72 + c * 32 + 8 * half),
                    *(const v8bf*)(Pw + lm * 72 + c * 32 + 16 + 8 * half));
#pragma unroll
    for (int d = 0; d < 4; ++d) {
#pragma unroll
      for (int c = 0; c < 2; ++c) {
        const bf16_t* bp = VTs[cur] + (d * 16 + lm) * 72 + c * 32 + 8 * half;
        v16bf bfrag = join8(*(const v8bf*)bp, *(const v8bf*)(bp + 16));
        o[d] = __builtin_amdgcn_wmma_f32_16x16x32_bf16(
            false, pa[c], false, bfrag, (short)0, o[d], false, false);
      }
    }
  }

  // ctx[B*S, D] bf16, heads re-interleaved
#pragma unroll
  for (int r = 0; r < 8; ++r) {
    float inv = 1.0f / l_run[r];
    size_t grow = (size_t)b * SEQ + qbase + r + 8 * half;
#pragma unroll
    for (int d = 0; d < 4; ++d)
      ctx[grow * D_MODEL + h * D_K_ + d * 16 + lm] = (bf16_t)(o[d][r] * inv);
  }
}

// ---------------------------------------------------------------------------
// out = LN(x + gattn + 0.3*(depthwise conv3(x) + cb)); one block per token.
// ---------------------------------------------------------------------------
__global__ __launch_bounds__(256) void fuse_conv_ln_kernel(
    const float* __restrict__ x, const float* __restrict__ gattn,
    const float* __restrict__ conv_w, const float* __restrict__ conv_b,
    const float* __restrict__ ln_g, const float* __restrict__ ln_b,
    float* __restrict__ out)
{
  const int row = blockIdx.x;            // b*SEQ + s
  const int s   = row & (SEQ - 1);
  const int tid = threadIdx.x;
  const size_t base = (size_t)row * D_MODEL;
  float y[4];
  float s1 = 0.f, s2 = 0.f;
#pragma unroll
  for (int j = 0; j < 4; ++j) {
    int d = tid * 4 + j;
    float xm = (s > 0)       ? x[base - D_MODEL + d] : 0.f;
    float x0 = x[base + d];
    float xp = (s < SEQ - 1) ? x[base + D_MODEL + d] : 0.f;
    float local = xm * conv_w[d * 3 + 0] + x0 * conv_w[d * 3 + 1]
                + xp * conv_w[d * 3 + 2] + conv_b[d];
    float v = x0 + gattn[base + d] + 0.3f * local;
    y[j] = v; s1 += v; s2 += v * v;
  }
#pragma unroll
  for (int off = 16; off > 0; off >>= 1) {
    s1 += __shfl_xor(s1, off, 32);
    s2 += __shfl_xor(s2, off, 32);
  }
  __shared__ float red[2][8];
  int wave = tid >> 5, lane = tid & 31;
  if (lane == 0) { red[0][wave] = s1; red[1][wave] = s2; }
  __syncthreads();
  float ts1 = 0.f, ts2 = 0.f;
#pragma unroll
  for (int w = 0; w < 8; ++w) { ts1 += red[0][w]; ts2 += red[1][w]; }
  float mu  = ts1 * (1.0f / D_MODEL);
  float var = ts2 * (1.0f / D_MODEL) - mu * mu;
  float rsq = rsqrtf(var + LN_EPS);
#pragma unroll
  for (int j = 0; j < 4; ++j) {
    int d = tid * 4 + j;
    out[base + d] = (y[j] - mu) * rsq * ln_g[d] + ln_b[d];
  }
}

// ---------------------------------------------------------------------------
extern "C" void kernel_launch(void* const* d_in, const int* in_sizes, int n_in,
                              void* d_out, int out_size, void* d_ws, size_t ws_size,
                              hipStream_t stream)
{
  (void)in_sizes; (void)n_in; (void)out_size; (void)ws_size;
  const float* x  = (const float*)d_in[0];
  const float* wq = (const float*)d_in[1];
  const float* bq = (const float*)d_in[2];
  const float* wk = (const float*)d_in[3];
  const float* bk = (const float*)d_in[4];
  const float* wv = (const float*)d_in[5];
  const float* bv = (const float*)d_in[6];
  const float* wo = (const float*)d_in[7];
  const float* bo = (const float*)d_in[8];
  const float* cw = (const float*)d_in[9];
  const float* cb = (const float*)d_in[10];
  const float* lg = (const float*)d_in[11];
  const float* lb = (const float*)d_in[12];

  // workspace layout (fixed offsets, ~120 MB total)
  char* ws = (char*)d_ws;
  const size_t MB = 1024u * 1024u;
  bf16_t* xb   = (bf16_t*)(ws + 0);        // 16 MB  x bf16 [8192,1024]
  bf16_t* wqb  = (bf16_t*)(ws + 16 * MB);  //  2 MB
  bf16_t* wkb  = (bf16_t*)(ws + 18 * MB);  //  2 MB
  bf16_t* wvb  = (bf16_t*)(ws + 20 * MB);  //  2 MB
  bf16_t* wob  = (bf16_t*)(ws + 22 * MB);  //  2 MB
  bf16_t* qb   = (bf16_t*)(ws + 24 * MB);  // 16 MB  q [B,H,S,Dk] (pre-scaled)
  bf16_t* kb   = (bf16_t*)(ws + 40 * MB);  // 16 MB  k [B,H,S,Dk]
  bf16_t* vb   = (bf16_t*)(ws + 56 * MB);  // 16 MB  v [B,H,Dk,S]
  bf16_t* ctxb = (bf16_t*)(ws + 72 * MB);  // 16 MB  ctx bf16 [8192,1024]
  float*  gat  = (float*) (ws + 88 * MB);  // 32 MB  gattn fp32 [8192,1024]

  const int Ntok = BATCH * SEQ;            // 8192

  f32_to_bf16_kernel<<<4096, 256, 0, stream>>>(x,  xb,  Ntok * D_MODEL);
  f32_to_bf16_kernel<<<1024, 256, 0, stream>>>(wq, wqb, D_MODEL * D_MODEL);
  f32_to_bf16_kernel<<<1024, 256, 0, stream>>>(wk, wkb, D_MODEL * D_MODEL);
  f32_to_bf16_kernel<<<1024, 256, 0, stream>>>(wv, wvb, D_MODEL * D_MODEL);
  f32_to_bf16_kernel<<<1024, 256, 0, stream>>>(wo, wob, D_MODEL * D_MODEL);

  dim3 gg(D_MODEL / 128, Ntok / 128);      // (8, 64)
  gemm_xwT_kernel<OUT_BHSD><<<gg, 256, 0, stream>>>(
      xb, wqb, bq, qb, Ntok, D_MODEL, D_MODEL, 0.125f);   // fold 1/sqrt(64)
  gemm_xwT_kernel<OUT_BHSD><<<gg, 256, 0, stream>>>(
      xb, wkb, bk, kb, Ntok, D_MODEL, D_MODEL, 1.0f);
  gemm_xwT_kernel<OUT_BHDS><<<gg, 256, 0, stream>>>(
      xb, wvb, bv, vb, Ntok, D_MODEL, D_MODEL, 1.0f);

  attn_kernel<<<dim3(SEQ / 128, NHEAD, BATCH), 256, 0, stream>>>(qb, kb, vb, ctxb);

  gemm_xwT_kernel<OUT_F32_RM><<<gg, 256, 0, stream>>>(
      ctxb, wob, bo, gat, Ntok, D_MODEL, D_MODEL, 1.0f);

  fuse_conv_ln_kernel<<<Ntok, 256, 0, stream>>>(x, gat, cw, cb, lg, lb, (float*)d_out);
}
